// Informer_RL_21904333209933
// MI455X (gfx1250) — compile-verified, matrix operand order checked by reading, not run on hardware
//
#include <hip/hip_runtime.h>
#include <cmath>

#define DEV __device__ __forceinline__

typedef __attribute__((ext_vector_type(16))) __bf16 v16bf;
typedef __attribute__((ext_vector_type(8)))  float  v8f;

constexpr int BATCH  = 64;
constexpr int ESTEPS = 336;
constexpr int DM     = 512;
constexpr int NH     = 8;
constexpr int DH     = 64;
constexpr int PRED   = 12;
constexpr int LABEL  = 168;
constexpr int HSTR   = 192;   // history row stride (>= LABEL + PRED - 1)

// ---------- helpers ----------

DEV __bf16 f2bf(float f) {
  union { float f; unsigned u; } a{f};
  unsigned u = a.u;
  u += 0x7fffu + ((u >> 16) & 1u);            // round-to-nearest-even
  union { unsigned short s; __bf16 b; } o{(unsigned short)(u >> 16)};
  return o.b;
}

DEV float gelu_f(float x) { return 0.5f * x * (1.0f + erff(x * 0.70710678118654752f)); }

// f32 -> bf16 elementwise (amortized: each bf16 row is read by 32 N-tiles)
__global__ void tobf16_k(const float* __restrict__ x, __bf16* __restrict__ y, int n) {
  const int i = blockIdx.x * 256 + threadIdx.x;
  if (i < n) y[i] = f2bf(x[i]);
}

// Load one lane's slice of a 16x32 bf16 WMMA operand (row-major, stride 512).
// ISA 16-bit A layout: lane<16 -> K = {0..7, 16..23}; lane>=16 -> K = {8..15, 24..31}
// (the +0/+8 base is folded into the pointer). Each 8-group is one b128 load.
DEV v16bf ldtile(const __bf16* p) {
  v16bf r;
#pragma unroll
  for (int i = 0; i < 8; ++i) r[i] = p[i];
#pragma unroll
  for (int i = 0; i < 8; ++i) r[8 + i] = p[16 + i];
  return r;
}

// ---------- GEMM: Y[M,512] = act(Xbf[M,512] @ Wbf[512,512]^T + bias (+resid)) ----------
// grid = (M/16, 8), block = 128 (4 waves); each wave owns a 16x16 tile.
__global__ void gemm_wmma_k(const __bf16* __restrict__ X, const __bf16* __restrict__ W,
                            const float* __restrict__ bias, const float* resid,
                            float* Y, int M, int act) {
  const int lane = threadIdx.x & 31;
  const int wave = threadIdx.x >> 5;
  const int m0 = blockIdx.x << 4;
  const int n0 = (blockIdx.y << 6) + (wave << 4);
  const int koff = (lane >> 4) << 3;                       // 0 or 8
  const __bf16* ap = X + (size_t)(m0 + (lane & 15)) * DM + koff;
  const __bf16* bp = W + (size_t)(n0 + (lane & 15)) * DM + koff;
  v8f c = {};
#pragma unroll 4
  for (int kk = 0; kk < DM; kk += 32) {
    if (kk + 128 < DM) {                                   // stay inside the row
      __builtin_prefetch(ap + kk + 128, 0, 3);             // -> global_prefetch_b8 (near)
      __builtin_prefetch(bp + kk + 128, 0, 3);
    }
    v16bf a = ldtile(ap + kk);
    v16bf b = ldtile(bp + kk);
    c = __builtin_amdgcn_wmma_f32_16x16x32_bf16(false, a, false, b,
                                                (short)0, c, false, false);
  }
  // C/D layout: lane&15 = N column, lanes>=16 start at M row 8.
  const int n  = n0 + (lane & 15);
  const int mb = m0 + ((lane >> 4) << 3);
  const float bv = bias[n];
#pragma unroll
  for (int r = 0; r < 8; ++r) {
    const size_t idx = (size_t)(mb + r) * DM + n;
    float v = c[r] + bv;
    if (resid) v += resid[idx];
    if (act)   v = gelu_f(v);
    Y[idx] = v;
  }
}

// ---------- LayerNorm over 512, one wave per row ----------
__global__ void ln_k(const float* X, const float* g, const float* b, float* Y, int M) {
  const int wave = threadIdx.x >> 5, lane = threadIdx.x & 31;
  const int row = (blockIdx.x << 2) + wave;
  if (row >= M) return;
  const float* x = X + (size_t)row * DM;
  float v[16], s = 0.f;
#pragma unroll
  for (int i = 0; i < 16; ++i) { v[i] = x[lane + (i << 5)]; s += v[i]; }
#pragma unroll
  for (int o = 16; o; o >>= 1) s += __shfl_xor(s, o, 32);
  const float mean = s * (1.f / DM);
  float s2 = 0.f;
#pragma unroll
  for (int i = 0; i < 16; ++i) { float d = v[i] - mean; s2 += d * d; }
#pragma unroll
  for (int o = 16; o; o >>= 1) s2 += __shfl_xor(s2, o, 32);
  const float rstd = rsqrtf(s2 * (1.f / DM) + 1e-5f);
  float* y = Y + (size_t)row * DM;
#pragma unroll
  for (int i = 0; i < 16; ++i) {
    const int c = lane + (i << 5);
    y[c] = (v[i] - mean) * rstd * g[c] + b[c];
  }
}

// ---------- token embedding: circular conv(k=3) + sinusoidal PE ----------
__global__ void embed_k(const float* x, int xstride, const float* w, float* out, int L) {
  const int d = threadIdx.x;           // 0..511
  const int t = blockIdx.x;            // token 0..B*L-1
  const int b = t / L, l = t - b * L;
  const float* xb = x + (size_t)b * xstride;
  float tok = xb[(l + L - 1) % L] * w[d * 3 + 0]
            + xb[l]               * w[d * 3 + 1]
            + xb[(l + 1) % L]     * w[d * 3 + 2];
  const int i2 = d >> 1;
  const float div = __expf((float)(i2 << 1) * (-9.210340371976184f / (float)DM));
  const float ang = (float)l * div;
  out[(size_t)t * DM + d] = tok + ((d & 1) ? __cosf(ang) : __sinf(ang));
}

// ---------- ProbSparse measure M = max_k(q.k) - mean_k(q.k), one wave per (b,h,l) ----------
__global__ void prob_m_k(const float* Qm, const float* Km, float* Msc,
                         int L, int LK, int total) {
  const int wave = threadIdx.x >> 5, lane = threadIdx.x & 31;
  const int wg = (blockIdx.x << 2) + wave;
  if (wg >= total) return;
  const int l = wg % L, h = (wg / L) & (NH - 1), b = wg / (L * NH);
  const float* q = Qm + (size_t)(b * L + l) * DM + h * DH;
  float qr[DH];
#pragma unroll
  for (int i = 0; i < DH; ++i) qr[i] = q[i];
  const float* Kb = Km + (size_t)b * LK * DM + h * DH;
  float mx = -1e30f, sm = 0.f;
  for (int k = lane; k < LK; k += 32) {
    const float* kp = Kb + (size_t)k * DM;
    float d = 0.f;
#pragma unroll
    for (int i = 0; i < DH; ++i) d += qr[i] * kp[i];
    mx = fmaxf(mx, d); sm += d;
  }
  for (int o = 16; o; o >>= 1) { mx = fmaxf(mx, __shfl_xor(mx, o, 32)); sm += __shfl_xor(sm, o, 32); }
  if (lane == 0) Msc[wg] = mx - sm / (float)LK;
}

// ---------- top-u selection per (b,h): O(u*L) wave argmax ----------
__global__ void topk_k(const float* Msc, int* top, int L, int u) {
  const int bh = blockIdx.x, lane = threadIdx.x;
  const float* m = Msc + (size_t)bh * L;
  unsigned used = 0;                                      // bit i: index lane+32*i taken
  for (int it = 0; it < u; ++it) {
    float best = -1e30f; int bidx = 0;
    for (int i = 0, idx = lane; idx < L; ++i, idx += 32)
      if (!((used >> i) & 1u)) { float v = m[idx]; if (v > best) { best = v; bidx = idx; } }
    for (int o = 16; o; o >>= 1) {
      float ov = __shfl_xor(best, o, 32); int oi = __shfl_xor(bidx, o, 32);
      if (ov > best) { best = ov; bidx = oi; }
    }
    bidx = __shfl(bidx, 0, 32);                           // make selection wave-uniform
    if ((bidx & 31) == lane) used |= 1u << (bidx >> 5);
    if (lane == 0) top[(size_t)bh * u + it] = bidx;
  }
}

// ---------- fill ctx (B,H,L,DH) with mean over keys of V ----------
__global__ void meanv_k(const float* Vm, float* ctx, int L) {
  const int bh = blockIdx.x, d = threadIdx.x;             // blockDim = 64
  const int b = bh >> 3, h = bh & 7;
  const float* Vb = Vm + (size_t)b * L * DM + h * DH + d;
  float s = 0.f;
  for (int k = 0; k < L; ++k) s += Vb[(size_t)k * DM];
  s /= (float)L;
  float* c = ctx + (size_t)bh * L * DH + d;
  for (int l = 0; l < L; ++l) c[(size_t)l * DH] = s;
}

// ---------- one-query online-softmax attention (wave32; lanes over head dims) ----------
DEV void attn_query(const float* q, const float* Kb, const float* Vb, int LK,
                    int lane, float& r0, float& r1) {
  const float q0 = q[lane], q1 = q[lane + 32];
  float m = -1e30f, ssum = 0.f, a0 = 0.f, a1 = 0.f;
  for (int k = 0; k < LK; ++k) {
    const float* kp = Kb + (size_t)k * DM;
    float p = q0 * kp[lane] + q1 * kp[lane + 32];
#pragma unroll
    for (int o = 16; o; o >>= 1) p += __shfl_xor(p, o, 32);
    p *= 0.125f;                                          // 1/sqrt(64)
    const float nm = fmaxf(m, p);
    const float cr = __expf(m - nm), w = __expf(p - nm);
    const float* vp = Vb + (size_t)k * DM;
    ssum = ssum * cr + w;
    a0 = a0 * cr + w * vp[lane];
    a1 = a1 * cr + w * vp[lane + 32];
    m = nm;
  }
  r0 = a0 / ssum; r1 = a1 / ssum;
}

// full attention for every query; writes directly into (B,LQ,H*DH) layout
__global__ void full_attn_k(const float* Qm, const float* Km, const float* Vm,
                            float* Om, int LQ, int LK, int total) {
  const int wave = threadIdx.x >> 5, lane = threadIdx.x & 31;
  const int wg = (blockIdx.x << 2) + wave;
  if (wg >= total) return;
  const int l = wg % LQ, h = (wg / LQ) & 7, b = wg / (LQ * 8);
  const float* q  = Qm + (size_t)(b * LQ + l) * DM + h * DH;
  const float* Kb = Km + (size_t)b * LK * DM + h * DH;
  const float* Vb = Vm + (size_t)b * LK * DM + h * DH;
  float r0, r1;
  attn_query(q, Kb, Vb, LK, lane, r0, r1);
  float* o = Om + (size_t)(b * LQ + l) * DM + h * DH;
  o[lane] = r0; o[lane + 32] = r1;
}

// attention only for the top-u queries; overwrites rows of ctx (B,H,L,DH)
__global__ void prob_top_k(const float* Qm, const float* Km, const float* Vm,
                           const int* top, float* ctx, int L, int u, int total) {
  const int wave = threadIdx.x >> 5, lane = threadIdx.x & 31;
  const int wg = (blockIdx.x << 2) + wave;
  if (wg >= total) return;
  const int ui = wg % u, bh = wg / u, b = bh >> 3, h = bh & 7;
  const int l = top[(size_t)bh * u + ui];
  const float* q  = Qm + (size_t)(b * L + l) * DM + h * DH;
  const float* Kb = Km + (size_t)b * L * DM + h * DH;
  const float* Vb = Vm + (size_t)b * L * DM + h * DH;
  float r0, r1;
  attn_query(q, Kb, Vb, L, lane, r0, r1);
  float* o = ctx + ((size_t)bh * L + l) * DH;
  o[lane] = r0; o[lane + 32] = r1;
}

// (B,H,L,E) -> (B,L,H*E)  (encoder, mix=False)
__global__ void ctx_unmix_k(const float* ctx, float* out, int L, int n) {
  const int i = blockIdx.x * 256 + threadIdx.x;
  if (i >= n) return;
  const int c = i & (DM - 1);
  const int t = i >> 9;
  const int l = t % L, b = t / L;
  const int h = c >> 6, e = c & 63;
  out[i] = ctx[(((size_t)(b * 8 + h) * L + l) * DH) + e];
}

// history init: last LABEL samples of input_x
__global__ void hist_init_k(const float* x, float* hist) {
  const int i = blockIdx.x * 256 + threadIdx.x;
  if (i >= BATCH * LABEL) return;
  const int b = i / LABEL, l = i - b * LABEL;
  hist[(size_t)b * HSTR + l] = x[(size_t)b * ESTEPS + (ESTEPS - LABEL) + l];
}

// projection of last token + write output + append to history
__global__ void proj_out_k(const float* D, int Lh, const float* pw, const float* pb,
                           float* hist, int step, float* out) {
  const int b = blockIdx.x, lane = threadIdx.x;
  const float* x = D + (size_t)(b * Lh + (Lh - 1)) * DM;
  float s = 0.f;
  for (int i = lane; i < DM; i += 32) s += x[i] * pw[i];
  for (int o = 16; o; o >>= 1) s += __shfl_xor(s, o, 32);
  if (lane == 0) {
    const float p = s + pb[0];
    out[b * PRED + step] = p;
    hist[(size_t)b * HSTR + LABEL + step] = p;
  }
}

// ---------- host ----------

static int prob_u(int L) {
  int v = 5 * (int)std::ceil(std::log((double)L));
  return v < L ? v : L;
}

extern "C" void kernel_launch(void* const* d_in, const int* in_sizes, int n_in,
                              void* d_out, int out_size, void* d_ws, size_t ws_size,
                              hipStream_t stream) {
  (void)in_sizes; (void)n_in; (void)out_size; (void)ws_size;
  const float* input_x     = (const float*)d_in[0];
  const float* enc_emb_w   = (const float*)d_in[1];
  const float* dec_emb_w   = (const float*)d_in[2];
  const float* enc_attn_w  = (const float*)d_in[3];
  const float* enc_attn_b  = (const float*)d_in[4];
  const float* enc_ff1_w   = (const float*)d_in[5];
  const float* enc_ff1_b   = (const float*)d_in[6];
  const float* enc_ff2_w   = (const float*)d_in[7];
  const float* enc_ff2_b   = (const float*)d_in[8];
  const float* enc_ln_g    = (const float*)d_in[9];
  const float* enc_ln_b    = (const float*)d_in[10];
  const float* enc_fg      = (const float*)d_in[11];
  const float* enc_fb      = (const float*)d_in[12];
  const float* dec_self_w  = (const float*)d_in[13];
  const float* dec_self_b  = (const float*)d_in[14];
  const float* dec_cross_w = (const float*)d_in[15];
  const float* dec_cross_b = (const float*)d_in[16];
  const float* dec_ff1_w   = (const float*)d_in[17];
  const float* dec_ff1_b   = (const float*)d_in[18];
  const float* dec_ff2_w   = (const float*)d_in[19];
  const float* dec_ff2_b   = (const float*)d_in[20];
  const float* dec_ln_g    = (const float*)d_in[21];
  const float* dec_ln_b    = (const float*)d_in[22];
  const float* dec_fg      = (const float*)d_in[23];
  const float* dec_fb      = (const float*)d_in[24];
  const float* proj_w      = (const float*)d_in[25];
  const float* proj_b      = (const float*)d_in[26];
  float* out = (float*)d_out;

  // workspace carve-out (~450 MB assumed available)
  float* ws = (float*)d_ws;
  size_t off = 0;
  auto carve = [&](size_t n) { float* p = ws + off; off += n; return p; };
  const size_t ET  = (size_t)BATCH * ESTEPS;   // 21504 tokens
  const size_t DT  = (size_t)BATCH * 180;      // max decoder tokens
  const size_t WSZ = (size_t)DM * DM;
  float *X  = carve(ET * DM), *Y  = carve(ET * DM), *Q  = carve(ET * DM);
  float *Kc = carve(ET * DM), *Vc = carve(ET * DM), *C  = carve(ET * DM);
  float *dX = carve(DT * DM), *dY = carve(DT * DM), *dQ = carve(DT * DM);
  float *dK = carve(DT * DM), *dV = carve(DT * DM), *dC = carve(DT * DM);
  float* Msc = carve((size_t)BATCH * NH * ESTEPS);
  int*   top = (int*)carve((size_t)BATCH * NH * 32);
  float* hist = carve((size_t)BATCH * HSTR);
  // bf16 staging: one activation buffer + all weights converted once per call
  __bf16* Abf = (__bf16*)carve(ET * DM / 2);
  __bf16* Wbf = (__bf16*)carve(22 * WSZ / 2);              // 22 x 512x512 bf16

  auto conv = [&](const float* src, __bf16* dst, int n) {
    tobf16_k<<<dim3((n + 255) / 256), dim3(256), 0, stream>>>(src, dst, n);
  };
  auto gemm = [&](const __bf16* Xp, const __bf16* Wp, const float* Bp,
                  const float* Rp, float* Yp, int M, int act) {
    gemm_wmma_k<<<dim3(M / 16, 8), dim3(128), 0, stream>>>(Xp, Wp, Bp, Rp, Yp, M, act);
  };
  auto ln = [&](const float* Xp, const float* Gp, const float* Bp, float* Yp, int M) {
    ln_k<<<dim3((M + 3) / 4), dim3(128), 0, stream>>>(Xp, Gp, Bp, Yp, M);
  };

  // ---- weight bf16 conversion (~11.5 MB total, L2-resident) ----
  __bf16* wb_enc_attn = Wbf;                 // 2*4 matrices
  __bf16* wb_enc_ff1  = wb_enc_attn + 8 * WSZ;   // 2
  __bf16* wb_enc_ff2  = wb_enc_ff1  + 2 * WSZ;   // 2
  __bf16* wb_self     = wb_enc_ff2  + 2 * WSZ;   // 4
  __bf16* wb_cross    = wb_self     + 4 * WSZ;   // 4
  __bf16* wb_dff1     = wb_cross    + 4 * WSZ;   // 1
  __bf16* wb_dff2     = wb_dff1     + 1 * WSZ;   // 1
  conv(enc_attn_w,  wb_enc_attn, 8 * WSZ);
  conv(enc_ff1_w,   wb_enc_ff1,  2 * WSZ);
  conv(enc_ff2_w,   wb_enc_ff2,  2 * WSZ);
  conv(dec_self_w,  wb_self,     4 * WSZ);
  conv(dec_cross_w, wb_cross,    4 * WSZ);
  conv(dec_ff1_w,   wb_dff1,     1 * WSZ);
  conv(dec_ff2_w,   wb_dff2,     1 * WSZ);

  // ===== encoder =====
  embed_k<<<dim3(BATCH * ESTEPS), dim3(DM), 0, stream>>>(input_x, ESTEPS, enc_emb_w, X, ESTEPS);
  const int ME = BATCH * ESTEPS;
  const int uE = prob_u(ESTEPS);                            // = 30
  for (int l = 0; l < 2; ++l) {
    const __bf16* w = wb_enc_attn + (size_t)l * 4 * WSZ;
    const float* bb = enc_attn_b + (size_t)l * 4 * DM;
    conv(X, Abf, ME * DM);                                  // one conversion feeds q,k,v
    gemm(Abf, w,           bb,          nullptr, Q,  ME, 0);
    gemm(Abf, w + WSZ,     bb + DM,     nullptr, Kc, ME, 0);
    gemm(Abf, w + 2 * WSZ, bb + 2 * DM, nullptr, Vc, ME, 0);
    const int tot = BATCH * NH * ESTEPS;
    prob_m_k<<<dim3((tot + 3) / 4), dim3(128), 0, stream>>>(Q, Kc, Msc, ESTEPS, ESTEPS, tot);
    topk_k<<<dim3(BATCH * NH), dim3(32), 0, stream>>>(Msc, top, ESTEPS, uE);
    meanv_k<<<dim3(BATCH * NH), dim3(64), 0, stream>>>(Vc, C, ESTEPS);
    const int tt = BATCH * NH * uE;
    prob_top_k<<<dim3((tt + 3) / 4), dim3(128), 0, stream>>>(Q, Kc, Vc, top, C, ESTEPS, uE, tt);
    const int nel = ME * DM;
    ctx_unmix_k<<<dim3((nel + 255) / 256), dim3(256), 0, stream>>>(C, Y, ESTEPS, nel);
    conv(Y, Abf, ME * DM);
    gemm(Abf, w + 3 * WSZ, bb + 3 * DM, X, X, ME, 0);
    ln(X, enc_ln_g + (size_t)l * 2 * DM,      enc_ln_b + (size_t)l * 2 * DM,      X, ME);
    conv(X, Abf, ME * DM);
    gemm(Abf, wb_enc_ff1 + (size_t)l * WSZ, enc_ff1_b + (size_t)l * DM, nullptr, Y, ME, 1);
    conv(Y, Abf, ME * DM);
    gemm(Abf, wb_enc_ff2 + (size_t)l * WSZ, enc_ff2_b + (size_t)l * DM, X, X, ME, 0);
    ln(X, enc_ln_g + (size_t)l * 2 * DM + DM, enc_ln_b + (size_t)l * 2 * DM + DM, X, ME);
  }
  ln(X, enc_fg, enc_fb, X, ME);                             // X = encoder output

  // cross-attention K/V depend only on encoder output: hoist out of the decode loop
  conv(X, Abf, ME * DM);
  gemm(Abf, wb_cross + WSZ,     dec_cross_b + DM,     nullptr, Kc, ME, 0);
  gemm(Abf, wb_cross + 2 * WSZ, dec_cross_b + 2 * DM, nullptr, Vc, ME, 0);

  hist_init_k<<<dim3((BATCH * LABEL + 255) / 256), dim3(256), 0, stream>>>(input_x, hist);

  // ===== autoregressive decoder =====
  for (int s = 0; s < PRED; ++s) {
    const int Lh = LABEL + s;                               // 168..179
    const int MD = BATCH * Lh;                              // multiple of 16
    const int uD = prob_u(Lh);                              // = 30
    embed_k<<<dim3(MD), dim3(DM), 0, stream>>>(hist, HSTR, dec_emb_w, dX, Lh);

    // prob self-attention (mix=True)
    conv(dX, Abf, MD * DM);
    gemm(Abf, wb_self,           dec_self_b,          nullptr, dQ, MD, 0);
    gemm(Abf, wb_self + WSZ,     dec_self_b + DM,     nullptr, dK, MD, 0);
    gemm(Abf, wb_self + 2 * WSZ, dec_self_b + 2 * DM, nullptr, dV, MD, 0);
    const int tot = BATCH * NH * Lh;
    prob_m_k<<<dim3((tot + 3) / 4), dim3(128), 0, stream>>>(dQ, dK, Msc, Lh, Lh, tot);
    topk_k<<<dim3(BATCH * NH), dim3(32), 0, stream>>>(Msc, top, Lh, uD);
    meanv_k<<<dim3(BATCH * NH), dim3(64), 0, stream>>>(dV, dC, Lh);
    const int tt = BATCH * NH * uD;
    prob_top_k<<<dim3((tt + 3) / 4), dim3(128), 0, stream>>>(dQ, dK, dV, top, dC, Lh, uD, tt);
    // mix=True: (B,H,L,E) flat IS (B,L,512) flat -> feed ctx straight into out-proj
    conv(dC, Abf, MD * DM);
    gemm(Abf, wb_self + 3 * WSZ, dec_self_b + 3 * DM, dX, dX, MD, 0);
    ln(dX, dec_ln_g, dec_ln_b, dX, MD);

    // full cross-attention against encoder K/V
    conv(dX, Abf, MD * DM);
    gemm(Abf, wb_cross, dec_cross_b, nullptr, dQ, MD, 0);
    const int ct = BATCH * NH * Lh;
    full_attn_k<<<dim3((ct + 3) / 4), dim3(128), 0, stream>>>(dQ, Kc, Vc, dY, Lh, ESTEPS, ct);
    conv(dY, Abf, MD * DM);
    gemm(Abf, wb_cross + 3 * WSZ, dec_cross_b + 3 * DM, dX, dX, MD, 0);
    ln(dX, dec_ln_g + DM, dec_ln_b + DM, dX, MD);

    // FFN
    conv(dX, Abf, MD * DM);
    gemm(Abf, wb_dff1, dec_ff1_b, nullptr, dY, MD, 1);
    conv(dY, Abf, MD * DM);
    gemm(Abf, wb_dff2, dec_ff2_b, dX, dX, MD, 0);
    ln(dX, dec_ln_g + 2 * DM, dec_ln_b + 2 * DM, dX, MD);
    ln(dX, dec_fg, dec_fb, dX, MD);

    proj_out_k<<<dim3(BATCH), dim3(32), 0, stream>>>(dX, Lh, proj_w, proj_b, hist, s, out);
  }
}
